// AdaptiveWeightedMultiGraphConstruction_41815801594316
// MI455X (gfx1250) — matrix operand order repaired and checked by reading, not compiled, not run on hardware
//
#include <hip/hip_runtime.h>
#include <hip/hip_bf16.h>

typedef __attribute__((ext_vector_type(8)))  _Float16 v8h;
typedef __attribute__((ext_vector_type(16))) _Float16 v16h;
typedef __attribute__((ext_vector_type(8)))  float    v8f;

#define NROWS 8192
#define NDIM  256
#define EPSN  1e-8f

// Phase 1: L2-normalize each row in fp32, emit f16 row-major copy (xn) and
// f16 transposed copy (xnT) (not read by cosine_wmma; its B-operand loads read xn).
__global__ void normalize_f16(const float* __restrict__ feat,
                              _Float16* __restrict__ xn,
                              _Float16* __restrict__ xnT) {
    const int row = blockIdx.x;
    const int tid = threadIdx.x;          // 256 threads == D
    const float v = feat[row * NDIM + tid];

    __shared__ float red[256];
    red[tid] = v * v;
    __syncthreads();
    #pragma unroll
    for (int s = 128; s > 0; s >>= 1) {
        if (tid < s) red[tid] += red[tid + s];
        __syncthreads();
    }
    const float inv = 1.0f / fmaxf(sqrtf(red[0]), EPSN);
    const _Float16 h = (_Float16)(v * inv);
    xn[row * NDIM + tid] = h;
    xnT[(size_t)tid * NROWS + row] = h;
}

// Phase 2: sim = clamp(xn @ xn^T, 0) via v_wmma_f32_16x16x32_f16.
// Each wave computes a 32x32 output block (2x2 tiles of 16x16) to halve
// operand traffic vs one-tile-per-wave. 8 waves per block arranged 2x4:
// block covers 64 rows x 128 cols of output.
__global__ void __launch_bounds__(256, 2)
cosine_wmma(const _Float16* __restrict__ xn,
            const _Float16* __restrict__ xnT,
            float* __restrict__ out) {
    const int lane = threadIdx.x & 31;
    const int wave = threadIdx.x >> 5;
    const int wr   = wave >> 2;            // 0..1
    const int wc   = wave & 3;             // 0..3

    const int i0 = blockIdx.y * 64  + wr * 32;   // rows  [i0, i0+32)
    const int j0 = blockIdx.x * 128 + wc * 32;   // cols  [j0, j0+32)

    // A-operand addressing (16-bit A 16x32 layout, ISA 7.12.2):
    //   lane<16:  M=lane,    K = {k0+0..7,  k0+16..23}
    //   lane>=16: M=lane-16, K = {k0+8..15, k0+24..31}
    const int rowA = i0 + (lane & 15);
    const int kHi  = (lane >> 4) << 3;                 // +0 or +8
    const _Float16* pa0 = xn + (size_t)rowA * NDIM + kHi;          // tile row 0
    const _Float16* pa1 = pa0 + (size_t)16 * NDIM;                 // tile row 1

    // B-operand addressing (16-bit B 32x16 layout):
    //   N = lane&15; K order same as A (table order).
    //   B[k][n] = xn[j0+n][k]  -> contiguous in xn.
    const _Float16* pb0 = xn + (size_t)(j0 + (lane & 15)) * NDIM + kHi; // tile col 0
    const _Float16* pb1 = pb0 + (size_t)16 * NDIM;                 // tile col 1

    v8f c00 = {}, c01 = {}, c10 = {}, c11 = {};

    #pragma unroll
    for (int k0 = 0; k0 < NDIM; k0 += 32) {
        union { v16h v; v8h h[2]; } a0, a1, b0, b1;
        a0.h[0] = *(const v8h*)(pa0 + k0);
        a0.h[1] = *(const v8h*)(pa0 + k0 + 16);
        a1.h[0] = *(const v8h*)(pa1 + k0);
        a1.h[1] = *(const v8h*)(pa1 + k0 + 16);

        const size_t kb = (size_t)k0 * NROWS;
        b0.h[0] = *(const v8h*)(pb0 + k0);
        b0.h[1] = *(const v8h*)(pb0 + k0 + 16);
        b1.h[0] = *(const v8h*)(pb1 + k0);
        b1.h[1] = *(const v8h*)(pb1 + k0 + 16);

        // 8 args: (neg_a, A, neg_b, B, c_mod, C, reuse_a, reuse_b)
        c00 = __builtin_amdgcn_wmma_f32_16x16x32_f16(
                  false, a0.v, false, b0.v, (short)0, c00, false, false);
        c01 = __builtin_amdgcn_wmma_f32_16x16x32_f16(
                  false, a0.v, false, b1.v, (short)0, c01, false, false);
        c10 = __builtin_amdgcn_wmma_f32_16x16x32_f16(
                  false, a1.v, false, b0.v, (short)0, c10, false, false);
        c11 = __builtin_amdgcn_wmma_f32_16x16x32_f16(
                  false, a1.v, false, b1.v, (short)0, c11, false, false);
    }

    // C/D layout: VGPR r -> M = r + 8*(lane>=16), N = lane&15.
    const int nlo   = lane & 15;
    const int mOff  = (lane >> 4) << 3;    // 0 or 8
    const size_t r0 = (size_t)(i0 + mOff) * NROWS;       // tile row 0 base
    const size_t r1 = (size_t)(i0 + 16 + mOff) * NROWS;  // tile row 1 base

    #pragma unroll
    for (int r = 0; r < 8; ++r) {
        const size_t rr0 = r0 + (size_t)r * NROWS;
        const size_t rr1 = r1 + (size_t)r * NROWS;
        out[rr0 + j0 + nlo]      = fmaxf(c00[r], 0.0f);
        out[rr0 + j0 + 16 + nlo] = fmaxf(c01[r], 0.0f);
        out[rr1 + j0 + nlo]      = fmaxf(c10[r], 0.0f);
        out[rr1 + j0 + 16 + nlo] = fmaxf(c11[r], 0.0f);
    }
}

extern "C" void kernel_launch(void* const* d_in, const int* in_sizes, int n_in,
                              void* d_out, int out_size, void* d_ws, size_t ws_size,
                              hipStream_t stream) {
    const float* feat = (const float*)d_in[0];
    float* out = (float*)d_out;

    // Workspace: xn (8192x256 f16, 4MB) then xnT (256x8192 f16, 4MB).
    _Float16* xn  = (_Float16*)d_ws;
    _Float16* xnT = xn + (size_t)NROWS * NDIM;

    normalize_f16<<<NROWS, 256, 0, stream>>>(feat, xn, xnT);

    dim3 grid(NROWS / 128, NROWS / 64);   // (64, 128): 64x128 output per block
    cosine_wmma<<<grid, 256, 0, stream>>>(xn, xnT, out);
}
